// AttractorGenerator_12747462934699
// MI455X (gfx1250) — compile-verified, matrix operand order checked by reading, not run on hardware
//
#include <hip/hip_runtime.h>

// ---------------------------------------------------------------------------
// AttractorGenerator for MI455X (gfx1250, wave32, WMMA).
//  Prep    : Wctx/Wk/Wv fp32 -> f16, transposed [out][in] (1.5 MB, L2-resident)
//  Phase 1 : fused ctx/K/V GEMMs via v_wmma_f32_16x16x32_f16.
//            A tile in LDS (f16), B fragments loaded straight from L2 in WMMA
//            register layout (no LDS staging, no K-loop barriers).
//            K/V stored f16 (128 MB -> resident in 192 MB L2 for phase 2).
//  Phase 2 : 16 sequential steps: flash-style single-query MHA partials +
//            one-block finalize (softmax combine, Wo, GRU, heads).
// ---------------------------------------------------------------------------

typedef __attribute__((ext_vector_type(16))) _Float16 v16h;
typedef __attribute__((ext_vector_type(8)))  _Float16 v8h;
typedef __attribute__((ext_vector_type(8)))  float    v8f;

constexpr int kHdim  = 512;   // H
constexpr int kAttr  = 256;
constexpr int kHeads = 8;
constexpr int kHd    = 64;    // head dim
constexpr int kSteps = 16;    // num_attractors (K_MAX)
constexpr int kMtile = 64;    // rows per workgroup in phase 1 (4 row-tiles)
constexpr int kLda   = 520;   // LDS row stride (halves): 1040B = 260 dw ≡ 4 (mod 64 banks)
constexpr int kChunk = 128;   // attention rows per block in phase 2

// ---------------- wave helpers (wave32) ----------------
__device__ __forceinline__ float wsum(float x) {
#pragma unroll
  for (int o = 16; o; o >>= 1) x += __shfl_xor(x, o);
  return x;
}
__device__ __forceinline__ float wmax(float x) {
#pragma unroll
  for (int o = 16; o; o >>= 1) x = fmaxf(x, __shfl_xor(x, o));
  return x;
}
__device__ __forceinline__ float sigmoidf_(float x) { return 1.0f / (1.0f + __expf(-x)); }
__device__ __forceinline__ v8f vzero8() {
  v8f z = {0.f, 0.f, 0.f, 0.f, 0.f, 0.f, 0.f, 0.f};
  return z;
}

// Load one 16x32 f16 WMMA operand fragment (LDS or global source).
// CDNA5 ISA 16-bit A layout: lane L<16 -> row M=L, K = 0..7 (V0-3) and
// 16..23 (V4-7); lane L>=16 -> row M=L-16, K = 8..15 and 24..31.
// => two contiguous 16-byte loads per lane. B uses the symmetric layout
// (N as "row"), sourced from the pre-transposed f16 WT[out][in] matrix.
__device__ __forceinline__ v16h ld_frag(const _Float16* base, int ld) {
  const int lane = threadIdx.x & 31;
  const _Float16* p = base + (lane & 15) * ld + ((lane >> 4) << 3);
  v8h lo = *(const v8h*)(p);
  v8h hi = *(const v8h*)(p + 16);
  return __builtin_shufflevector(lo, hi, 0, 1, 2, 3, 4, 5, 6, 7, 8, 9, 10, 11, 12, 13, 14, 15);
}

__device__ __forceinline__ v8f wmma16(v16h a, v16h b, v8f c) {
  // D = A(16x32 f16) * B(32x16 f16) + C(16x16 f32)
  return __builtin_amdgcn_wmma_f32_16x16x32_f16(false, a, false, b, (short)0, c, false, false);
}

// acc[4][4] += lAsrc(64x512 f16 LDS, ld=kLda) @ WT^T, where WT is f16 [512 out][512 in].
// Wave w owns output columns [w*64, w*64+64). No LDS writes -> no barriers.
__device__ __forceinline__ void gemm_block(const _Float16* lAsrc, const _Float16* __restrict__ WT,
                                           v8f acc[4][4], int w) {
#pragma unroll 1
  for (int kk = 0; kk < kHdim; kk += 32) {
    v16h bf[4];
#pragma unroll
    for (int ct = 0; ct < 4; ++ct)
      bf[ct] = ld_frag(WT + (size_t)(w * 64 + ct * 16) * kHdim + kk, kHdim);
#pragma unroll
    for (int rt = 0; rt < 4; ++rt) {
      v16h af = ld_frag(lAsrc + rt * 16 * kLda + kk, kLda);
#pragma unroll
      for (int ct = 0; ct < 4; ++ct) acc[rt][ct] = wmma16(af, bf[ct], acc[rt][ct]);
    }
  }
}

// ---------------- prep: W fp32 [in][out] -> f16 WT [out][in] ----------------
__global__ __launch_bounds__(256) void convert_w(const float* __restrict__ W,
                                                 _Float16* __restrict__ WT) {
  const int e = blockIdx.x * 256 + threadIdx.x;  // 512*512 / 256 = 1024 blocks
  const int k = e >> 9;
  const int c = e & (kHdim - 1);
  WT[(size_t)c * kHdim + k] = (_Float16)W[(size_t)k * kHdim + c];
}

// ---------------- Phase 1: fused ctx -> (colsum, K16, V16) ----------------
__global__ __launch_bounds__(256) void fused_ctx_kv(
    const float* __restrict__ A, const _Float16* __restrict__ wt_ctx,
    const float* __restrict__ bctx,
    const _Float16* __restrict__ wt_k, const float* __restrict__ bk,
    const _Float16* __restrict__ wt_v, const float* __restrict__ bv,
    _Float16* __restrict__ k16, _Float16* __restrict__ v16, float* __restrict__ colsum) {
  extern __shared__ char smem[];
  _Float16* lA   = (_Float16*)smem;                         // 64x520 f16 (later: store staging)
  _Float16* lctx = (_Float16*)(smem + kMtile * kLda * 2);   // 64x520 f16

  const int tid  = threadIdx.x;
  const int w    = tid >> 5;
  const int lane = tid & 31;
  const size_t row0 = (size_t)blockIdx.x * kMtile;

  // frames tile fp32 -> f16 LDS (coalesced along c)
  for (int e = tid; e < kMtile * kHdim; e += 256) {
    const int r = e >> 9, c = e & (kHdim - 1);
    lA[r * kLda + c] = (_Float16)A[(row0 + r) * kHdim + c];
  }
  __syncthreads();

  // ---- ctx = A @ Wctx + b_ctx ----
  v8f acc[4][4];
#pragma unroll
  for (int rt = 0; rt < 4; ++rt)
#pragma unroll
    for (int ct = 0; ct < 4; ++ct) acc[rt][ct] = vzero8();
  gemm_block(lA, wt_ctx, acc, w);

  // epilogue: bias, per-column partial sums (for h0), ctx -> f16 LDS
#pragma unroll
  for (int ct = 0; ct < 4; ++ct) {
    const int col = w * 64 + ct * 16 + (lane & 15);
    const float b = bctx[col];
    float part = 0.f;
#pragma unroll
    for (int rt = 0; rt < 4; ++rt) {
#pragma unroll
      for (int i = 0; i < 8; ++i) {
        const int r = rt * 16 + ((lane >> 4) << 3) + i;
        const float v = acc[rt][ct][i] + b;
        part += v;
        lctx[r * kLda + col] = (_Float16)v;
      }
    }
    part += __shfl_xor(part, 16);  // combine the two half-wave row sets
    if (lane < 16) colsum[(size_t)blockIdx.x * kHdim + col] = part;
  }
  __syncthreads();

  // ---- K = ctx @ Wk + bk ; V = ctx @ Wv + bv (f16, staged for b128 stores) ----
  const _Float16* WT2[2] = {wt_k, wt_v};
  const float* b2[2]     = {bk, bv};
  _Float16* o2[2]        = {k16, v16};
  _Float16* stage        = lA;  // frames tile no longer needed
#pragma unroll 1
  for (int t = 0; t < 2; ++t) {
    v8f a2[4][4];
#pragma unroll
    for (int rt = 0; rt < 4; ++rt)
#pragma unroll
      for (int ct = 0; ct < 4; ++ct) a2[rt][ct] = vzero8();
    gemm_block(lctx, WT2[t], a2, w);
#pragma unroll
    for (int ct = 0; ct < 4; ++ct) {
      const int col = w * 64 + ct * 16 + (lane & 15);
      const float b = b2[t][col];
#pragma unroll
      for (int rt = 0; rt < 4; ++rt)
#pragma unroll
        for (int i = 0; i < 8; ++i) {
          const int r = rt * 16 + ((lane >> 4) << 3) + i;
          stage[r * kHdim + col] = (_Float16)(a2[rt][ct][i] + b);
        }
    }
    __syncthreads();
    // contiguous 64x512 f16 block -> coalesced b128 stores
    const uint4* src = (const uint4*)stage;
    uint4* dst = (uint4*)(o2[t] + row0 * kHdim);
    for (int e = tid; e < kMtile * kHdim / 8; e += 256) dst[e] = src[e];
    __syncthreads();
  }
}

// ---------------- init: h0 = mean(ctx), q0 = h0@Wq+bq, prev_a = start ----------------
__global__ __launch_bounds__(256) void init_state(
    const float* __restrict__ colsum, int nblk, float invN,
    const float* __restrict__ start_token, const float* __restrict__ Wq,
    const float* __restrict__ bq, float* __restrict__ hbuf, float* __restrict__ qbuf,
    float* __restrict__ pabuf) {
  __shared__ float h0[kHdim];
  const int tid = threadIdx.x;
#pragma unroll
  for (int u = 0; u < 2; ++u) {
    const int j = tid + u * 256;
    float s = 0.f;
    for (int b = 0; b < nblk; ++b) s += colsum[(size_t)b * kHdim + j];
    const float hj = s * invN;
    h0[j] = hj;
    hbuf[j] = hj;
  }
  pabuf[tid] = start_token[tid];
  __syncthreads();
#pragma unroll
  for (int u = 0; u < 2; ++u) {
    const int j = tid + u * 256;
    float a = bq[j];
    for (int i = 0; i < kHdim; ++i) a += h0[i] * Wq[(size_t)i * kHdim + j];
    qbuf[j] = a;
  }
}

// ---------------- Phase 2a: per-chunk online-softmax partials ----------------
__global__ __launch_bounds__(256) void attn_partial(
    const _Float16* __restrict__ k16, const _Float16* __restrict__ v16,
    const float* __restrict__ qv, float* __restrict__ part) {
  const int w = threadIdx.x >> 5;  // wave == head
  const int lane = threadIdx.x & 31;
  float q[kHd];
#pragma unroll
  for (int d = 0; d < kHd; ++d) q[d] = qv[w * kHd + d];
  float m = -3.0e38f, s = 0.0f;
  float ctx[kHd];
#pragma unroll
  for (int d = 0; d < kHd; ++d) ctx[d] = 0.0f;

  const int n0 = blockIdx.x * kChunk;
#pragma unroll 1
  for (int it = 0; it < kChunk / 32; ++it) {
    const int n = n0 + it * 32 + lane;
    const v8h* kp = (const v8h*)(k16 + ((size_t)n * kHdim + w * kHd));
    const v8h* vp = (const v8h*)(v16 + ((size_t)n * kHdim + w * kHd));
    if (it + 1 < kChunk / 32) {  // K/V are L2-resident; pull next lines toward L0
      __builtin_prefetch((const void*)(k16 + ((size_t)(n + 32) * kHdim + w * kHd)), 0, 1);
      __builtin_prefetch((const void*)(v16 + ((size_t)(n + 32) * kHdim + w * kHd)), 0, 1);
    }
    float dot = 0.0f;
#pragma unroll
    for (int j = 0; j < 8; ++j) {
      v8h kvv = kp[j];
#pragma unroll
      for (int e = 0; e < 8; ++e) dot += (float)kvv[e] * q[j * 8 + e];
    }
    const float logit = dot * 0.125f;  // 1/sqrt(64)
    const float mn = fmaxf(m, logit);
    const float corr = __expf(m - mn);
    const float p = __expf(logit - mn);
    s = s * corr + p;
#pragma unroll
    for (int j = 0; j < 8; ++j) {
      v8h vvv = vp[j];
#pragma unroll
      for (int e = 0; e < 8; ++e) ctx[j * 8 + e] = ctx[j * 8 + e] * corr + p * (float)vvv[e];
    }
    m = mn;
  }
  const float M = wmax(m);
  const float f = __expf(m - M);
  const float sl = wsum(s * f);
  float* dst = part + ((size_t)blockIdx.x * kHeads + w) * 66;
#pragma unroll
  for (int d = 0; d < kHd; ++d) {
    const float cd = wsum(ctx[d] * f);
    if (lane == 0) dst[d] = cd;
  }
  if (lane == 0) { dst[64] = M; dst[65] = sl; }
}

// ---------------- Phase 2b: combine + Wo + GRU + heads + next q ----------------
__global__ __launch_bounds__(256) void attn_finalize(
    const float* __restrict__ part, int nch, const float* __restrict__ Wo,
    const float* __restrict__ bo, const float* __restrict__ Wih,
    const float* __restrict__ Whh, const float* __restrict__ bih,
    const float* __restrict__ bhn, const float* __restrict__ Wattr,
    const float* __restrict__ battr, const float* __restrict__ Wconf,
    const float* __restrict__ bconf, const float* __restrict__ Wq,
    const float* __restrict__ bq, float* __restrict__ hbuf, float* __restrict__ qbuf,
    float* __restrict__ pabuf, float* __restrict__ out, int step) {
  __shared__ float hv[kHdim], ctxv[kHdim], att[kHdim], xv[kAttr + kHdim];
  __shared__ float ihv[3 * kHdim], hhv[3 * kHdim], hn[kHdim];
  const int tid = threadIdx.x, w = tid >> 5, lane = tid & 31;

  hv[tid] = hbuf[tid];
  hv[tid + 256] = hbuf[tid + 256];
  xv[tid] = pabuf[tid];  // prev attractor (ATTR = 256)
  __syncthreads();

  // per-head softmax combine across chunks (wave w == head w)
  {
    float m = -3.0e38f;
    for (int c = lane; c < nch; c += 32) m = fmaxf(m, part[((size_t)c * kHeads + w) * 66 + 64]);
    const float M = wmax(m);
    float s = 0.f;
    for (int c = lane; c < nch; c += 32) {
      const float* pc = part + ((size_t)c * kHeads + w) * 66;
      s += pc[65] * __expf(pc[64] - M);
    }
    const float S = wsum(s);
    float a0 = 0.f, a1 = 0.f;
    for (int c = 0; c < nch; ++c) {
      const float* pc = part + ((size_t)c * kHeads + w) * 66;
      const float e = __expf(pc[64] - M);
      a0 += pc[lane] * e;
      a1 += pc[lane + 32] * e;
    }
    ctxv[w * kHd + lane] = a0 / S;
    ctxv[w * kHd + lane + 32] = a1 / S;
  }
  __syncthreads();

  // att = ctxv @ Wo + bo
  {
    const int j0 = tid, j1 = tid + 256;
    float a0 = bo[j0], a1 = bo[j1];
    for (int i = 0; i < kHdim; ++i) {
      const float ci = ctxv[i];
      a0 += ci * Wo[(size_t)i * kHdim + j0];
      a1 += ci * Wo[(size_t)i * kHdim + j1];
    }
    att[j0] = a0;
    att[j1] = a1;
  }
  __syncthreads();
  xv[kAttr + tid] = att[tid];
  xv[kAttr + 256 + tid] = att[tid + 256];
  __syncthreads();

  // GRU matvecs: ih = Wih@x + bih ; hh = Whh@h   (wave-dot per row, coalesced)
  for (int r = w; r < 3 * kHdim; r += 8) {
    float p1 = 0.f;
    for (int i = lane; i < kAttr + kHdim; i += 32) p1 += Wih[(size_t)r * (kAttr + kHdim) + i] * xv[i];
    p1 = wsum(p1);
    if (lane == 0) ihv[r] = p1 + bih[r];
    float p2 = 0.f;
    for (int i = lane; i < kHdim; i += 32) p2 += Whh[(size_t)r * kHdim + i] * hv[i];
    p2 = wsum(p2);
    if (lane == 0) hhv[r] = p2;
  }
  __syncthreads();

  // gates
#pragma unroll
  for (int u = 0; u < 2; ++u) {
    const int e = tid + u * 256;
    const float rg = sigmoidf_(ihv[e] + hhv[e]);
    const float zg = sigmoidf_(ihv[kHdim + e] + hhv[kHdim + e]);
    const float ng = tanhf(ihv[2 * kHdim + e] + rg * (hhv[2 * kHdim + e] + bhn[e]));
    const float hne = (1.f - zg) * ng + zg * hv[e];
    hn[e] = hne;
    hbuf[e] = hne;
  }
  __syncthreads();

  // attractor head + confidence
  for (int r = w; r < kAttr; r += 8) {
    float p = 0.f;
    for (int i = lane; i < kHdim; i += 32) p += Wattr[(size_t)r * kHdim + i] * hn[i];
    p = wsum(p);
    if (lane == 0) {
      const float av = p + battr[r];
      out[step * kAttr + r] = av;
      pabuf[r] = av;
    }
  }
  if (w == 0) {
    float p = 0.f;
    for (int i = lane; i < kHdim; i += 32) p += Wconf[i] * hn[i];
    p = wsum(p);
    if (lane == 0) out[kSteps * kAttr + step] = sigmoidf_(p + bconf[0]);
  }

  // q_next = h_new @ Wq + bq
  {
    const int j0 = tid, j1 = tid + 256;
    float a0 = bq[j0], a1 = bq[j1];
    for (int i = 0; i < kHdim; ++i) {
      const float hi = hn[i];
      a0 += hi * Wq[(size_t)i * kHdim + j0];
      a1 += hi * Wq[(size_t)i * kHdim + j1];
    }
    qbuf[j0] = a0;
    qbuf[j1] = a1;
  }
}

// ---------------------------------------------------------------------------
extern "C" void kernel_launch(void* const* d_in, const int* in_sizes, int n_in,
                              void* d_out, int out_size, void* d_ws, size_t ws_size,
                              hipStream_t stream) {
  const float* frames = (const float*)d_in[0];
  const float* Wctx   = (const float*)d_in[1];
  const float* bctx   = (const float*)d_in[2];
  const float* Wq     = (const float*)d_in[3];
  const float* bq     = (const float*)d_in[4];
  const float* Wk     = (const float*)d_in[5];
  const float* bk     = (const float*)d_in[6];
  const float* Wv     = (const float*)d_in[7];
  const float* bv     = (const float*)d_in[8];
  const float* Wo     = (const float*)d_in[9];
  const float* bo     = (const float*)d_in[10];
  const float* Wih    = (const float*)d_in[11];
  const float* Whh    = (const float*)d_in[12];
  const float* bih    = (const float*)d_in[13];
  const float* bhn    = (const float*)d_in[14];
  const float* start_token = (const float*)d_in[15];
  const float* Wattr  = (const float*)d_in[16];
  const float* battr  = (const float*)d_in[17];
  const float* Wconf  = (const float*)d_in[18];
  const float* bconf  = (const float*)d_in[19];
  float* out = (float*)d_out;

  const int N    = in_sizes[0] / kHdim;  // 65536
  const int nblk = N / kMtile;           // 1024
  const int nch  = N / kChunk;           // 512

  // workspace carve-out (~133.7 MB)
  char* ws = (char*)d_ws;
  size_t off = 0;
  _Float16* k16 = (_Float16*)(ws + off); off += (size_t)N * kHdim * 2;
  _Float16* v16 = (_Float16*)(ws + off); off += (size_t)N * kHdim * 2;
  _Float16* wt_ctx = (_Float16*)(ws + off); off += (size_t)kHdim * kHdim * 2;
  _Float16* wt_k   = (_Float16*)(ws + off); off += (size_t)kHdim * kHdim * 2;
  _Float16* wt_v   = (_Float16*)(ws + off); off += (size_t)kHdim * kHdim * 2;
  float* colsum = (float*)(ws + off);    off += (size_t)nblk * kHdim * 4;
  float* partb  = (float*)(ws + off);    off += (size_t)nch * kHeads * 66 * 4;
  float* hbuf   = (float*)(ws + off);    off += 4096;
  float* qbuf   = (float*)(ws + off);    off += 4096;
  float* pabuf  = (float*)(ws + off);    off += 4096;
  (void)ws_size; (void)n_in; (void)out_size;

  const int wblk = (kHdim * kHdim) / 256;  // 1024
  convert_w<<<wblk, 256, 0, stream>>>(Wctx, wt_ctx);
  convert_w<<<wblk, 256, 0, stream>>>(Wk, wt_k);
  convert_w<<<wblk, 256, 0, stream>>>(Wv, wt_v);

  const size_t smem1 = (size_t)kMtile * kLda * 2 * 2;  // 133,120 B (< 320 KB WGP)
  hipFuncSetAttribute((const void*)fused_ctx_kv,
                      hipFuncAttributeMaxDynamicSharedMemorySize, (int)smem1);
  fused_ctx_kv<<<nblk, 256, smem1, stream>>>(frames, wt_ctx, bctx, wt_k, bk, wt_v, bv,
                                             k16, v16, colsum);
  init_state<<<1, 256, 0, stream>>>(colsum, nblk, 1.0f / (float)N, start_token, Wq, bq,
                                    hbuf, qbuf, pabuf);
  for (int step = 0; step < kSteps; ++step) {
    attn_partial<<<nch, 256, 0, stream>>>(k16, v16, qbuf, partb);
    attn_finalize<<<1, 256, 0, stream>>>(partb, nch, Wo, bo, Wih, Whh, bih, bhn,
                                         Wattr, battr, Wconf, bconf, Wq, bq,
                                         hbuf, qbuf, pabuf, out, step);
  }
}
